// DecoderInitWrapper_51745765982789
// MI455X (gfx1250) — compile-verified, matrix operand order checked by reading, not run on hardware
//
#include <hip/hip_runtime.h>

// ---------------- model constants ----------------
constexpr int L_ = 4, D_ = 1024, HQ_ = 16, HK_ = 8, HD_ = 128, FF_ = 3072, V_ = 32000;
constexpr int B_ = 2, S_ = 1024;
constexpr int GROUPS_ = HQ_ / HK_;
constexpr int MBS = B_ * S_;               // 2048 token rows
constexpr float EPS_ = 1e-6f;
constexpr float THETA_ = 1000000.0f;
constexpr float SCALE_ = 0.08838834764831845f;  // 128^-0.5

// ---------------- types ----------------
typedef __bf16 bf16x16 __attribute__((ext_vector_type(16)));
typedef float  f32x8   __attribute__((ext_vector_type(8)));
typedef float  f32x4   __attribute__((ext_vector_type(4)));
typedef unsigned int u32x4 __attribute__((ext_vector_type(4)));

union FragU { u32x4 q[2]; bf16x16 v; };

__device__ __forceinline__ unsigned short f2bf(float f) {
  unsigned u = __builtin_bit_cast(unsigned, f);
  u += 0x7fffu + ((u >> 16) & 1u);            // round-to-nearest-even
  return (unsigned short)(u >> 16);
}
__device__ __forceinline__ unsigned pack2(float lo, float hi) {
  return (unsigned)f2bf(lo) | ((unsigned)f2bf(hi) << 16);
}

// ---------------- WMMA GEMM ----------------
// C[M,N] = A[M,K] @ op(B) (+ Res), bf16 inputs via LDS staging, f32 accum.
// Block tile 128x64, BK=32. 8 waves in 4(M) x 2(N); each wave owns a 32x32
// output (2 A-frags x 2 B-frags -> 4 WMMAs per K-step; 8 ds_load_b128).
// OPB==0: B is [K,N] row-major.  OPB==1: B is [N,K] row-major (A@B^T).
// mode 0: linear batch strides sA/sB/sC.
// mode 1: attention scores  (A=q [B,HQ,S,HD], B=k [B,HK,S,HD] w/ GQA, C=[BH,S,S])
// mode 2: attention output  (A=probs [BH,S,S], B=v [B,HK,S,HD] w/ GQA, C=[BH,S,HD])
constexpr int BM = 128, BN = 64, BK = 32;

template <int OPB, bool RES>
__global__ void __launch_bounds__(256)
gemm_wmma_kernel(const float* __restrict__ A, const float* __restrict__ Bm,
                 const float* __restrict__ Res, float* __restrict__ C,
                 int K, int lda, int ldb, int ldc, int mode,
                 long long sA, long long sB, long long sC) {
  __shared__ unsigned short As[BM * BK];     // [m][k]  8 KB
  __shared__ unsigned short Bs[BN * BK];     // [n][k]  4 KB (N-major)

  const int z = blockIdx.z;
  long long offA, offB, offC;
  if (mode == 1) {
    int b = z / HQ_, hh = z % HQ_;
    offA = (long long)z * S_ * HD_;
    offB = ((long long)b * HK_ + hh / GROUPS_) * S_ * HD_;
    offC = (long long)z * S_ * S_;
  } else if (mode == 2) {
    int b = z / HQ_, hh = z % HQ_;
    offA = (long long)z * S_ * S_;
    offB = ((long long)b * HK_ + hh / GROUPS_) * S_ * HD_;
    offC = (long long)z * S_ * HD_;
  } else {
    offA = (long long)z * sA; offB = (long long)z * sB; offC = (long long)z * sC;
  }
  A += offA; Bm += offB; C += offC;
  if (RES) Res += offC;

  const int t = threadIdx.x;
  const int tileM = blockIdx.y * BM;
  const int tileN = blockIdx.x * BN;

  const int wave = t >> 5;
  const int lane = t & 31;
  const int lrow = lane & 15;
  const int kq   = lane >> 4;                // 0: K 0-7/16-23, 1: K 8-15/24-31
  const int mw   = (wave >> 1) * 32;         // wave M origin (0,32,64,96)
  const int nw   = (wave & 1) * 32;          // wave N origin (0,32)

  f32x8 c00 = {0.f,0.f,0.f,0.f,0.f,0.f,0.f,0.f};
  f32x8 c01 = c00, c10 = c00, c11 = c00;

  // per-thread staging coordinates
  const int ar = t >> 2;                     // 0..63 (rows ar and ar+64)
  const int ac = (t & 3) * 8;                // 0,8,16,24
  const int kp = t >> 4;                     // 0..15 (k pair) for OPB==0
  const int nb = (t & 15) * 4;               // 0..60  for OPB==0
  const int bn = t >> 2;                     // 0..63  for OPB==1
  const int bk = (t & 3) * 8;                // 0,8,16,24 for OPB==1

  for (int k0 = 0; k0 < K; k0 += BK) {
    // ---- stage A tile (128x32): packed b128 stores ----
    {
      const float* p0 = A + (long long)(tileM + ar) * lda + (k0 + ac);
      const float* p1 = p0 + (long long)64 * lda;
      f32x4 a0 = *(const f32x4*)p0;
      f32x4 a1 = *(const f32x4*)(p0 + 4);
      f32x4 a2 = *(const f32x4*)p1;
      f32x4 a3 = *(const f32x4*)(p1 + 4);
      __builtin_prefetch(p0 + BK, 0, 1);     // next K-tile -> global_prefetch_b8
      __builtin_prefetch(p1 + BK, 0, 1);
      u32x4 w0 = {pack2(a0.x, a0.y), pack2(a0.z, a0.w), pack2(a1.x, a1.y), pack2(a1.z, a1.w)};
      u32x4 w1 = {pack2(a2.x, a2.y), pack2(a2.z, a2.w), pack2(a3.x, a3.y), pack2(a3.z, a3.w)};
      *(u32x4*)&As[ar * BK + ac] = w0;
      *(u32x4*)&As[(ar + 64) * BK + ac] = w1;
    }
    // ---- stage B tile (N-major LDS) ----
    if (OPB == 0) {                          // B[K,N]: two K-rows, packed b32 stores
      const float* p0 = Bm + (long long)(k0 + 2 * kp) * ldb + (tileN + nb);
      const float* p1 = p0 + ldb;
      f32x4 r0 = *(const f32x4*)p0;
      f32x4 r1 = *(const f32x4*)p1;
      __builtin_prefetch(p0 + (long long)BK * ldb, 0, 1);
      unsigned* bsw = (unsigned*)Bs;
      const int kh = kp;                     // u32 index along K (= k/2)
      bsw[(nb + 0) * (BK / 2) + kh] = pack2(r0.x, r1.x);
      bsw[(nb + 1) * (BK / 2) + kh] = pack2(r0.y, r1.y);
      bsw[(nb + 2) * (BK / 2) + kh] = pack2(r0.z, r1.z);
      bsw[(nb + 3) * (BK / 2) + kh] = pack2(r0.w, r1.w);
    } else {                                 // B[N,K]: packed b128 store
      const float* p = Bm + (long long)(tileN + bn) * ldb + (k0 + bk);
      f32x4 b0 = *(const f32x4*)p;
      f32x4 b1 = *(const f32x4*)(p + 4);
      __builtin_prefetch(p + BK, 0, 1);
      u32x4 w = {pack2(b0.x, b0.y), pack2(b0.z, b0.w), pack2(b1.x, b1.y), pack2(b1.z, b1.w)};
      *(u32x4*)&Bs[bn * BK + bk] = w;
    }
    __syncthreads();

    // ---- fragments per ISA 16-bit layout (ds_load_b128 x2 each) ----
    FragU a0, a1, b0, b1;
    const u32x4* Ar0 = (const u32x4*)&As[(mw + lrow) * BK];
    const u32x4* Ar1 = (const u32x4*)&As[(mw + 16 + lrow) * BK];
    a0.q[0] = Ar0[kq]; a0.q[1] = Ar0[2 + kq];
    a1.q[0] = Ar1[kq]; a1.q[1] = Ar1[2 + kq];
    const u32x4* Br0 = (const u32x4*)&Bs[(nw + lrow) * BK];
    const u32x4* Br1 = (const u32x4*)&Bs[(nw + 16 + lrow) * BK];
    b0.q[0] = Br0[kq]; b0.q[1] = Br0[2 + kq];
    b1.q[0] = Br1[kq]; b1.q[1] = Br1[2 + kq];

    c00 = __builtin_amdgcn_wmma_f32_16x16x32_bf16(false, a0.v, false, b0.v, (short)0, c00, false, false);
    c01 = __builtin_amdgcn_wmma_f32_16x16x32_bf16(false, a0.v, false, b1.v, (short)0, c01, false, false);
    c10 = __builtin_amdgcn_wmma_f32_16x16x32_bf16(false, a1.v, false, b0.v, (short)0, c10, false, false);
    c11 = __builtin_amdgcn_wmma_f32_16x16x32_bf16(false, a1.v, false, b1.v, (short)0, c11, false, false);
    __syncthreads();
  }

  // ---- epilogue: C/D layout: VGPR j -> row +j (+8 upper half-wave), col = lane%16 ----
  const int rb0 = tileM + mw + kq * 8;
  const int rb1 = rb0 + 16;
  const int cc0 = tileN + nw + lrow;
#pragma unroll
  for (int j = 0; j < 8; ++j) {
    long long i00 = (long long)(rb0 + j) * ldc + cc0;
    long long i10 = (long long)(rb1 + j) * ldc + cc0;
    float v00 = c00[j], v01 = c01[j], v10 = c10[j], v11 = c11[j];
    if (RES) {
      v00 += Res[i00]; v01 += Res[i00 + 16];
      v10 += Res[i10]; v11 += Res[i10 + 16];
    }
    C[i00] = v00; C[i00 + 16] = v01;
    C[i10] = v10; C[i10 + 16] = v11;
  }
}

// ---------------- RMSNorm over a row of length n ----------------
__global__ void __launch_bounds__(256)
rmsnorm_kernel(const float* __restrict__ x, const float* __restrict__ w,
               float* __restrict__ o, int n) {
  __shared__ float red[256];
  const long long row = blockIdx.x;
  const float* xr = x + row * n;
  float* orow = o + row * n;
  const int t = threadIdx.x;
  float s = 0.f;
  for (int i = t; i < n; i += 256) { float v = xr[i]; s += v * v; }
  red[t] = s; __syncthreads();
  for (int k = 128; k > 0; k >>= 1) { if (t < k) red[t] += red[t + k]; __syncthreads(); }
  const float inv = rsqrtf(red[0] / (float)n + EPS_);
  for (int i = t; i < n; i += 256) orow[i] = xr[i] * inv * w[i];
}

// -------- per-head RMSNorm + RoPE; [B*S, NH*HD] -> [B, NH, S, HD] --------
__global__ void __launch_bounds__(128)
rope_rms_kernel(const float* __restrict__ xin, const float* __restrict__ w,
                const int* __restrict__ pos, float* __restrict__ xout, int NH) {
  __shared__ float xs[HD_];
  __shared__ float red[HD_];
  const int bs = blockIdx.x, hh = blockIdx.y, d = threadIdx.x;
  const float v = xin[((long long)bs * NH + hh) * HD_ + d];
  red[d] = v * v; __syncthreads();
  for (int k = 64; k > 0; k >>= 1) { if (d < k) red[d] += red[d + k]; __syncthreads(); }
  const float xn = v * rsqrtf(red[0] / (float)HD_ + EPS_) * w[d];
  xs[d] = xn; __syncthreads();
  const int p = pos[bs];
  const int fi = d & 63;
  const float freq = (float)p * __expf(-((float)(2 * fi) / (float)HD_) * __logf(THETA_));
  const float c = __cosf(freq), sn = __sinf(freq);
  const float other = (d < 64) ? -xs[d + 64] : xs[d - 64];
  const float r = xn * c + other * sn;
  const int b = bs / S_, si = bs % S_;
  xout[(((long long)b * NH + hh) * S_ + si) * HD_ + d] = r;
}

// -------- V transpose: [B*S, HK*HD] -> [B, HK, S, HD] --------
__global__ void __launch_bounds__(128)
vtrans_kernel(const float* __restrict__ xin, float* __restrict__ xout) {
  const int bs = blockIdx.x, hh = blockIdx.y, d = threadIdx.x;
  const int b = bs / S_, si = bs % S_;
  xout[(((long long)b * HK_ + hh) * S_ + si) * HD_ + d] =
      xin[((long long)bs * HK_ + hh) * HD_ + d];
}

// -------- head transpose: [B, HQ, S, HD] -> [B*S, HQ*HD] --------
__global__ void __launch_bounds__(128)
otrans_kernel(const float* __restrict__ xin, float* __restrict__ xout) {
  const int bs = blockIdx.x, hh = blockIdx.y, d = threadIdx.x;
  const int b = bs / S_, si = bs % S_;
  xout[((long long)bs * HQ_ + hh) * HD_ + d] =
      xin[(((long long)b * HQ_ + hh) * S_ + si) * HD_ + d];
}

// -------- causal softmax over scores [BH, S, S], in place, with scale --------
__global__ void __launch_bounds__(256)
softmax_kernel(float* __restrict__ sc) {
  __shared__ float red[256];
  const long long rowid = blockIdx.x;
  const int r = (int)(rowid % S_);
  float* row = sc + rowid * S_;
  const int t = threadIdx.x;
  float mx = -3.4e38f;
  for (int c = t; c <= r; c += 256) mx = fmaxf(mx, row[c] * SCALE_);
  red[t] = mx; __syncthreads();
  for (int k = 128; k > 0; k >>= 1) { if (t < k) red[t] = fmaxf(red[t], red[t + k]); __syncthreads(); }
  mx = red[0]; __syncthreads();
  float sum = 0.f;
  for (int c = t; c < S_; c += 256) {
    float e = (c <= r) ? __expf(row[c] * SCALE_ - mx) : 0.f;
    row[c] = e; sum += e;
  }
  red[t] = sum; __syncthreads();
  for (int k = 128; k > 0; k >>= 1) { if (t < k) red[t] += red[t + k]; __syncthreads(); }
  const float inv = 1.f / red[0];
  for (int c = t; c < S_; c += 256) row[c] *= inv;
}

// -------- g = silu(g) * u --------
__global__ void __launch_bounds__(256)
silumul_kernel(float* __restrict__ g, const float* __restrict__ u, long long n) {
  const long long i = (long long)blockIdx.x * 256 + threadIdx.x;
  if (i < n) { float x = g[i]; g[i] = (x / (1.f + __expf(-x))) * u[i]; }
}

// ---------------- host driver ----------------
extern "C" void kernel_launch(void* const* d_in, const int* in_sizes, int n_in,
                              void* d_out, int out_size, void* d_ws, size_t ws_size,
                              hipStream_t stream) {
  const float* input_embeds = (const float*)d_in[0];
  const int*   pos  = (const int*)d_in[1];
  const float* Wq   = (const float*)d_in[2];
  const float* Wk   = (const float*)d_in[3];
  const float* Wv   = (const float*)d_in[4];
  const float* Wo   = (const float*)d_in[5];
  const float* qnw  = (const float*)d_in[6];
  const float* knw  = (const float*)d_in[7];
  const float* ln1  = (const float*)d_in[8];
  const float* ln2  = (const float*)d_in[9];
  const float* Wg   = (const float*)d_in[10];
  const float* Wu   = (const float*)d_in[11];
  const float* Wd   = (const float*)d_in[12];
  const float* nw   = (const float*)d_in[13];
  const float* lmh  = (const float*)d_in[14];
  float* out = (float*)d_out;

  // workspace layout (floats)
  float* ws     = (float*)d_ws;
  float* h      = ws;                                   //  2M  [B*S, D]
  float* nrm    = h      + (long long)MBS * D_;         //  2M  [B*S, D]
  float* qlin   = nrm    + (long long)MBS * D_;         //  4M  [B*S, HQ*HD] (also ohead)
  float* qrope  = qlin   + (long long)MBS * HQ_ * HD_;  //  4M  [B, HQ, S, HD] (also oflat)
  float* klin   = qrope  + (long long)MBS * HQ_ * HD_;  //  2M  [B*S, HK*HD]
  float* vlin   = klin   + (long long)MBS * HK_ * HD_;  //  2M  [B*S, HK*HD]
  float* scores = vlin   + (long long)MBS * HK_ * HD_;  // 32M  [B*HQ, S, S]
  float* gateb  = scores + (long long)B_ * HQ_ * S_ * S_; // 6M [B*S, FF]
  float* upb    = gateb  + (long long)MBS * FF_;          // 6M [B*S, FF]

  const long long LOGITS = (long long)MBS * V_;
  const long long KVSZ   = (long long)B_ * HK_ * S_ * HD_;
  float* keys_base = out + LOGITS;
  float* vals_base = out + LOGITS + (long long)L_ * KVSZ;

  hipMemcpyAsync(h, input_embeds, sizeof(float) * (size_t)MBS * D_,
                 hipMemcpyDeviceToDevice, stream);

  const dim3 blk(256);
  for (int l = 0; l < L_; ++l) {
    const float* Wq_l = Wq + (long long)l * D_ * HQ_ * HD_;
    const float* Wk_l = Wk + (long long)l * D_ * HK_ * HD_;
    const float* Wv_l = Wv + (long long)l * D_ * HK_ * HD_;
    const float* Wo_l = Wo + (long long)l * HQ_ * HD_ * D_;
    const float* Wg_l = Wg + (long long)l * D_ * FF_;
    const float* Wu_l = Wu + (long long)l * D_ * FF_;
    const float* Wd_l = Wd + (long long)l * FF_ * D_;
    float* kout = keys_base + (long long)l * KVSZ;
    float* vout = vals_base + (long long)l * KVSZ;

    // attn block
    rmsnorm_kernel<<<MBS, blk, 0, stream>>>(h, ln1 + (long long)l * D_, nrm, D_);
    gemm_wmma_kernel<0, false><<<dim3(HQ_ * HD_ / BN, MBS / BM, 1), blk, 0, stream>>>(
        nrm, Wq_l, nullptr, qlin, D_, D_, HQ_ * HD_, HQ_ * HD_, 0, 0, 0, 0);
    gemm_wmma_kernel<0, false><<<dim3(HK_ * HD_ / BN, MBS / BM, 1), blk, 0, stream>>>(
        nrm, Wk_l, nullptr, klin, D_, D_, HK_ * HD_, HK_ * HD_, 0, 0, 0, 0);
    gemm_wmma_kernel<0, false><<<dim3(HK_ * HD_ / BN, MBS / BM, 1), blk, 0, stream>>>(
        nrm, Wv_l, nullptr, vlin, D_, D_, HK_ * HD_, HK_ * HD_, 0, 0, 0, 0);
    rope_rms_kernel<<<dim3(MBS, HQ_), 128, 0, stream>>>(qlin, qnw + (long long)l * HD_, pos, qrope, HQ_);
    rope_rms_kernel<<<dim3(MBS, HK_), 128, 0, stream>>>(klin, knw + (long long)l * HD_, pos, kout, HK_);
    vtrans_kernel<<<dim3(MBS, HK_), 128, 0, stream>>>(vlin, vout);
    // scores = q @ k^T  (batched over B*HQ, GQA mapping inside)
    gemm_wmma_kernel<1, false><<<dim3(S_ / BN, S_ / BM, B_ * HQ_), blk, 0, stream>>>(
        qrope, kout, nullptr, scores, HD_, HD_, HD_, S_, 1, 0, 0, 0);
    softmax_kernel<<<B_ * HQ_ * S_, blk, 0, stream>>>(scores);
    // o = probs @ v  -> [B, HQ, S, HD] (reuse qlin)
    gemm_wmma_kernel<0, false><<<dim3(HD_ / BN, S_ / BM, B_ * HQ_), blk, 0, stream>>>(
        scores, vout, nullptr, qlin, S_, S_, HD_, HD_, 2, 0, 0, 0);
    otrans_kernel<<<dim3(MBS, HQ_), 128, 0, stream>>>(qlin, qrope);
    // h += o @ Wo
    gemm_wmma_kernel<0, true><<<dim3(D_ / BN, MBS / BM, 1), blk, 0, stream>>>(
        qrope, Wo_l, h, h, HQ_ * HD_, HQ_ * HD_, D_, D_, 0, 0, 0, 0);

    // MLP block
    rmsnorm_kernel<<<MBS, blk, 0, stream>>>(h, ln2 + (long long)l * D_, nrm, D_);
    gemm_wmma_kernel<0, false><<<dim3(FF_ / BN, MBS / BM, 1), blk, 0, stream>>>(
        nrm, Wg_l, nullptr, gateb, D_, D_, FF_, FF_, 0, 0, 0, 0);
    gemm_wmma_kernel<0, false><<<dim3(FF_ / BN, MBS / BM, 1), blk, 0, stream>>>(
        nrm, Wu_l, nullptr, upb, D_, D_, FF_, FF_, 0, 0, 0, 0);
    silumul_kernel<<<(unsigned)(((long long)MBS * FF_ + 255) / 256), blk, 0, stream>>>(
        gateb, upb, (long long)MBS * FF_);
    gemm_wmma_kernel<0, true><<<dim3(D_ / BN, MBS / BM, 1), blk, 0, stream>>>(
        gateb, Wd_l, h, h, FF_, FF_, D_, D_, 0, 0, 0, 0);
  }

  // final norm + lm_head -> logits
  rmsnorm_kernel<<<MBS, blk, 0, stream>>>(h, nw, nrm, D_);
  gemm_wmma_kernel<0, false><<<dim3(V_ / BN, MBS / BM, 1), blk, 0, stream>>>(
      nrm, lmh, nullptr, out, D_, D_, V_, V_, 0, 0, 0, 0);
}
